// MixingNetwork_12807592476838
// MI455X (gfx1250) — compile-verified, matrix operand order checked by reading, not run on hardware
//
#include <hip/hip_runtime.h>
#include <hip/hip_bf16.h>

typedef __bf16 bf16_t;
typedef __attribute__((ext_vector_type(16))) __bf16 v16bf;
typedef __attribute__((ext_vector_type(8)))  float  v8f;

#define NNODES_C 50000
#define NEDGES_C 800000
#define NGROUPS_C 10000
#define H_C 128
#define DOUT_C 64

// ---------------------------------------------------------------------------
// Fragment packing (CDNA5 ISA 7.12.2, 16-bit A 16x32 / B 32x16 layouts):
//   lane 0-15  : i<8 -> K = i,      i>=8 -> K = 16 + (i-8)
//   lane 16-31 : i<8 -> K = 8 + i,  i>=8 -> K = 24 + (i-8)
// A packed index: ((mt*Kt + kt)*32 + lane)*16 + i   (row = mt*16 + (lane&15))
// B packed index: ((kt*Nt + nt)*32 + lane)*16 + i   (col = nt*16 + (lane&15))
// Each lane fragment is a contiguous, 32B-aligned v16bf.
// ---------------------------------------------------------------------------
__device__ __forceinline__ int frag_k(int lane, int i) {
  int hi8 = (lane >> 4) * 8;
  return (i < 8) ? (hi8 + i) : (16 + hi8 + (i - 8));
}

__global__ void pack_a_bf16(const float* __restrict__ in, bf16_t* __restrict__ out,
                            int Mt, int Kt, int K) {
  long long total = (long long)Mt * Kt * 512;
  long long idx = (long long)blockIdx.x * blockDim.x + threadIdx.x;
  long long stride = (long long)gridDim.x * blockDim.x;
  for (; idx < total; idx += stride) {
    int i    = (int)(idx & 15);
    int lane = (int)((idx >> 4) & 31);
    long long t = idx >> 9;
    int kt = (int)(t % Kt);
    int mt = (int)(t / Kt);
    int row = mt * 16 + (lane & 15);
    int k   = kt * 32 + frag_k(lane, i);
    out[idx] = (bf16_t)in[(size_t)row * K + k];
  }
}

__global__ void pack_b_bf16(const float* __restrict__ in, bf16_t* __restrict__ out,
                            int Kt, int Nt, int Nn) {
  long long total = (long long)Kt * Nt * 512;
  long long idx = (long long)blockIdx.x * blockDim.x + threadIdx.x;
  long long stride = (long long)gridDim.x * blockDim.x;
  for (; idx < total; idx += stride) {
    int i    = (int)(idx & 15);
    int lane = (int)((idx >> 4) & 31);
    long long t = idx >> 9;
    int nt = (int)(t % Nt);
    int kt = (int)(t / Nt);
    int col = nt * 16 + (lane & 15);
    int k   = kt * 32 + frag_k(lane, i);
    out[idx] = (bf16_t)in[(size_t)k * Nn + col];
  }
}

// ---------------------------------------------------------------------------
// WMMA GEMM on fragment-packed operands. One wave = one 16-row strip x all
// NT column tiles; A fragment loaded once per K-step, then per column tile a
// 2x global_load_b128 clause feeding v_wmma_f32_16x16x32_bf16.
// ---------------------------------------------------------------------------
template <int NT>
__global__ void gemm_bf16_wmma_packed(const bf16_t* __restrict__ Ap,
                                      const bf16_t* __restrict__ Bp,
                                      const float*  __restrict__ Cin,
                                      float*        __restrict__ D,
                                      int Mt, int Kt) {
  const int Nn   = NT * 16;
  const int lane = threadIdx.x & 31;
  const int wave = threadIdx.x >> 5;
  const int tm   = blockIdx.x * (blockDim.x >> 5) + wave;
  if (tm >= Mt) return;  // uniform per wave; surviving waves keep EXEC all-ones

  const int l16 = lane & 15;
  const int hi8 = (lane >> 4) * 8;

  v8f acc[NT];
  if (Cin != nullptr) {
#pragma unroll
    for (int nt = 0; nt < NT; ++nt)
#pragma unroll
      for (int i = 0; i < 8; ++i)
        acc[nt][i] = Cin[(size_t)(tm * 16 + i + hi8) * Nn + nt * 16 + l16];
  } else {
#pragma unroll
    for (int nt = 0; nt < NT; ++nt)
#pragma unroll
      for (int i = 0; i < 8; ++i) acc[nt][i] = 0.0f;
  }

  const v16bf* Afrag = (const v16bf*)Ap + (size_t)tm * Kt * 32;
  const v16bf* Bfrag = (const v16bf*)Bp;
  for (int kt = 0; kt < Kt; ++kt) {
    v16bf a = Afrag[(size_t)kt * 32 + lane];
    if (kt + 1 < Kt)
      __builtin_prefetch((const void*)&Afrag[(size_t)(kt + 1) * 32 + lane], 0, 1);
    const v16bf* Bk = Bfrag + (size_t)kt * NT * 32 + lane;
#pragma unroll
    for (int nt = 0; nt < NT; ++nt) {
      v16bf b = Bk[nt * 32];
      acc[nt] = __builtin_amdgcn_wmma_f32_16x16x32_bf16(
          false, a, false, b, (short)0, acc[nt], false, false);
    }
  }

#pragma unroll
  for (int nt = 0; nt < NT; ++nt)
#pragma unroll
    for (int i = 0; i < 8; ++i)
      D[(size_t)(tm * 16 + i + hi8) * Nn + nt * 16 + l16] = acc[nt][i];
}

// ---------------------------------------------------------------------------
__global__ void edge_precompute(const float* __restrict__ ev,
                                const float* __restrict__ fa0, const float* __restrict__ fb0,
                                const float* __restrict__ fa1, const float* __restrict__ fb1,
                                const float* __restrict__ fa2, const float* __restrict__ fb2,
                                const float* __restrict__ fa3, const float* __restrict__ fb3,
                                float* __restrict__ s_all, int E) {
  int e = blockIdx.x * blockDim.x + threadIdx.x;
  if (e >= E) return;

  float vx = ev[3 * e + 0], vy = ev[3 * e + 1], vz = ev[3 * e + 2];
  float r = sqrtf(vx * vx + vy * vy + vz * vz);
  float inv = 1.0f / (r + 1e-12f);
  float X = vx * inv, Y = vy * inv, Z = vz * inv;

  const float s3 = 1.7320508075688772f;
  const float s5 = 2.2360679774997896f;
  const float s15 = 3.872983346207417f;
  float sh[9];
  sh[0] = 1.0f;
  sh[1] = s3 * X;  sh[2] = s3 * Y;  sh[3] = s3 * Z;
  sh[4] = s15 * X * Y;  sh[5] = s15 * Y * Z;
  sh[6] = 0.5f * s5 * (3.0f * Z * Z - 1.0f);
  sh[7] = s15 * X * Z;
  sh[8] = 0.5f * s15 * (X * X - Y * Y);

  float emb[10];
  const float step = 5.0f / 9.0f;
  const float sq10 = 3.1622776601683795f;
#pragma unroll
  for (int j = 0; j < 10; ++j) {
    float d = (r - (float)j * step) / step;
    emb[j] = (fabsf(d) < 1.0f) ? __cosf(1.5707963267948966f * d) * sq10 : 0.0f;
  }

  const float* FA[4] = {fa0, fa1, fa2, fa3};
  const float* FB[4] = {fb0, fb1, fb2, fb3};
  for (int l = 0; l < 4; ++l) {
    float w9[9] = {0, 0, 0, 0, 0, 0, 0, 0, 0};
    const float* fap = FA[l];
    const float* fbp = FB[l];
    for (int q = 0; q < 100; ++q) {
      float t = 0.0f;
#pragma unroll
      for (int j = 0; j < 10; ++j) t += emb[j] * fap[j * 100 + q];
      t = t / (1.0f + __expf(-t));  // silu
#pragma unroll
      for (int k = 0; k < 9; ++k) w9[k] += t * fbp[q * 9 + k];
    }
    float s = 0.0f;
#pragma unroll
    for (int k = 0; k < 9; ++k) s += w9[k] * sh[k];
    s_all[(size_t)l * E + e] = s;
  }
}

__global__ void edge_scatter(const float* __restrict__ xsn,
                             const int* __restrict__ src,
                             const int* __restrict__ dst,
                             const float* __restrict__ s,
                             float* __restrict__ agg,
                             long long total, int Hd) {
  long long idx = (long long)blockIdx.x * blockDim.x + threadIdx.x;
  long long stride = (long long)gridDim.x * blockDim.x;
  for (; idx < total; idx += stride) {
    int e = (int)(idx / Hd);
    int c = (int)(idx % Hd);
    float v = xsn[(size_t)src[e] * Hd + c] * (s[e] * 0.25f);  // 1/sqrt(16)
    atomicAdd(&agg[(size_t)dst[e] * Hd + c], v);
  }
}

__global__ void zero_f32(float* __restrict__ p, long long n) {
  long long idx = (long long)blockIdx.x * blockDim.x + threadIdx.x;
  long long stride = (long long)gridDim.x * blockDim.x;
  for (; idx < n; idx += stride) p[idx] = 0.0f;
}

__global__ void count_groups(const int* __restrict__ gidx, float* __restrict__ gcnt, int Nn) {
  int n = blockIdx.x * blockDim.x + threadIdx.x;
  if (n < Nn) atomicAdd(&gcnt[gidx[n]], 1.0f);
}

__global__ void silu_group_accum(const float* __restrict__ D,
                                 float* __restrict__ hnext,
                                 const int* __restrict__ gidx,
                                 float* __restrict__ gsum,
                                 int Nn, int Hd, int ld) {
  long long total = (long long)Nn * Hd;
  long long idx = (long long)blockIdx.x * blockDim.x + threadIdx.x;
  long long stride = (long long)gridDim.x * blockDim.x;
  for (; idx < total; idx += stride) {
    int n = (int)(idx / Hd);
    int c = (int)(idx % Hd);
    float v = D[(size_t)n * Hd + c];
    v = v / (1.0f + __expf(-v));
    hnext[(size_t)n * ld + c] = v;
    atomicAdd(&gsum[(size_t)gidx[n] * Hd + c], v);
  }
}

__global__ void group_mean_write(const float* __restrict__ gsum,
                                 const float* __restrict__ gcnt,
                                 const int* __restrict__ gidx,
                                 float* __restrict__ hnext,
                                 int Nn, int Hd, int ld) {
  long long total = (long long)Nn * Hd;
  long long idx = (long long)blockIdx.x * blockDim.x + threadIdx.x;
  long long stride = (long long)gridDim.x * blockDim.x;
  for (; idx < total; idx += stride) {
    int n = (int)(idx / Hd);
    int c = (int)(idx % Hd);
    int g = gidx[n];
    hnext[(size_t)n * ld + Hd + c] = gsum[(size_t)g * Hd + c] / fmaxf(gcnt[g], 1.0f);
  }
}

// ---------------------------------------------------------------------------
extern "C" void kernel_launch(void* const* d_in, const int* in_sizes, int n_in,
                              void* d_out, int out_size, void* d_ws, size_t ws_size,
                              hipStream_t stream) {
  (void)in_sizes; (void)n_in; (void)out_size; (void)ws_size;

  const int N = NNODES_C, E = NEDGES_C, G = NGROUPS_C;
  const int Mt = N / 16;  // 3125

  const float* x        = (const float*)d_in[0];
  const float* edge_vec = (const float*)d_in[1];
  const int*   esrc     = (const int*)d_in[2];
  const int*   edst     = (const int*)d_in[3];
  const int*   gidx     = (const int*)d_in[4];
  const float* Wsc[4] = {(const float*)d_in[5],  (const float*)d_in[10],
                         (const float*)d_in[15], (const float*)d_in[20]};
  const float* W1[4]  = {(const float*)d_in[6],  (const float*)d_in[11],
                         (const float*)d_in[16], (const float*)d_in[21]};
  const float* W2[4]  = {(const float*)d_in[7],  (const float*)d_in[12],
                         (const float*)d_in[17], (const float*)d_in[22]};
  const float* fa[4]  = {(const float*)d_in[8],  (const float*)d_in[13],
                         (const float*)d_in[18], (const float*)d_in[23]};
  const float* fb[4]  = {(const float*)d_in[9],  (const float*)d_in[14],
                         (const float*)d_in[19], (const float*)d_in[24]};

  const int dimsIn[4]  = {128, 256, 256, 256};
  const int dimsOut[4] = {128, 128, 128, 64};

  unsigned char* wsp = (unsigned char*)d_ws;
  size_t off = 0;
  auto carve = [&](size_t bytes) -> void* {
    void* p = wsp + off;
    off = (off + bytes + 255) & ~(size_t)255;
    return p;
  };
  float*  s_all  = (float*)carve((size_t)4 * E * sizeof(float));
  float*  hA     = (float*)carve((size_t)N * 256 * sizeof(float));
  float*  hB     = (float*)carve((size_t)N * 256 * sizeof(float));
  bf16_t* h_pk   = (bf16_t*)carve((size_t)N * 256 * sizeof(bf16_t));
  float*  sc     = (float*)carve((size_t)N * H_C * sizeof(float));
  float*  xsn    = (float*)carve((size_t)N * H_C * sizeof(float));
  float*  agg    = (float*)carve((size_t)N * H_C * sizeof(float));
  bf16_t* agg_pk = (bf16_t*)carve((size_t)N * H_C * sizeof(bf16_t));
  float*  gsum   = (float*)carve((size_t)G * H_C * sizeof(float));
  float*  gcnt   = (float*)carve((size_t)G * sizeof(float));
  bf16_t* WscP[4]; bf16_t* W1P[4]; bf16_t* W2P[4];
  for (int l = 0; l < 4; ++l) {
    WscP[l] = (bf16_t*)carve((size_t)dimsIn[l] * dimsOut[l] * sizeof(bf16_t));
    W1P[l]  = (bf16_t*)carve((size_t)dimsIn[l] * dimsOut[l] * sizeof(bf16_t));
    W2P[l]  = (bf16_t*)carve((size_t)dimsOut[l] * dimsOut[l] * sizeof(bf16_t));
  }

  auto zero = [&](float* p, long long n) {
    int blocks = (int)((n + 255) / 256); if (blocks > 8192) blocks = 8192;
    zero_f32<<<blocks, 256, 0, stream>>>(p, n);
  };
  auto packA = [&](const float* in, bf16_t* out, int K) {
    long long n = (long long)Mt * (K / 32) * 512;
    int blocks = (int)((n + 255) / 256); if (blocks > 8192) blocks = 8192;
    pack_a_bf16<<<blocks, 256, 0, stream>>>(in, out, Mt, K / 32, K);
  };
  auto packB = [&](const float* in, bf16_t* out, int K, int Nn) {
    long long n = (long long)(K / 32) * (Nn / 16) * 512;
    int blocks = (int)((n + 255) / 256); if (blocks > 4096) blocks = 4096;
    pack_b_bf16<<<blocks, 256, 0, stream>>>(in, out, K / 32, Nn / 16, Nn);
  };
  auto gemm = [&](const bf16_t* Ap, const bf16_t* Bp, const float* Cin, float* D,
                  int Nn, int K) {
    const int wavesPerBlock = 4;
    int blocks = (Mt + wavesPerBlock - 1) / wavesPerBlock;
    if (Nn == 128)
      gemm_bf16_wmma_packed<8><<<blocks, 32 * wavesPerBlock, 0, stream>>>(
          Ap, Bp, Cin, D, Mt, K / 32);
    else  // Nn == 64
      gemm_bf16_wmma_packed<4><<<blocks, 32 * wavesPerBlock, 0, stream>>>(
          Ap, Bp, Cin, D, Mt, K / 32);
  };

  edge_precompute<<<(E + 255) / 256, 256, 0, stream>>>(
      edge_vec, fa[0], fb[0], fa[1], fb[1], fa[2], fb[2], fa[3], fb[3], s_all, E);

  zero(gcnt, G);
  count_groups<<<(N + 255) / 256, 256, 0, stream>>>(gidx, gcnt, N);

  for (int l = 0; l < 4; ++l) {
    packB(Wsc[l], WscP[l], dimsIn[l], dimsOut[l]);
    packB(W1[l],  W1P[l],  dimsIn[l], dimsOut[l]);
    packB(W2[l],  W2P[l],  dimsOut[l], dimsOut[l]);
  }

  const float* hcur = x;
  int din = 128;
  float* hbufs[2] = {hA, hB};
  for (int l = 0; l < 3; ++l) {
    const int Ho = 128;
    packA(hcur, h_pk, din);
    gemm(h_pk, WscP[l], nullptr, sc,  Ho, din);      // sc  = h @ Wsc
    gemm(h_pk, W1P[l],  nullptr, xsn, Ho, din);      // xsn = h @ W1
    zero(agg, (long long)N * Ho);
    edge_scatter<<<8192, 256, 0, stream>>>(xsn, esrc, edst, s_all + (size_t)l * E,
                                           agg, (long long)E * Ho, Ho);
    packA(agg, agg_pk, Ho);
    gemm(agg_pk, W2P[l], sc, xsn, Ho, Ho);           // xsn = sc + agg @ W2
    zero(gsum, (long long)G * Ho);
    float* hnext = hbufs[l & 1];
    silu_group_accum<<<8192, 256, 0, stream>>>(xsn, hnext, gidx, gsum, N, Ho, 2 * Ho);
    group_mean_write<<<8192, 256, 0, stream>>>(gsum, gcnt, gidx, hnext, N, Ho, 2 * Ho);
    hcur = hnext;
    din = 256;
  }

  {
    const int Ho = DOUT_C;
    packA(hcur, h_pk, din);
    gemm(h_pk, WscP[3], nullptr, sc,  Ho, din);
    gemm(h_pk, W1P[3],  nullptr, xsn, Ho, din);
    zero(agg, (long long)N * Ho);
    edge_scatter<<<8192, 256, 0, stream>>>(xsn, esrc, edst, s_all + (size_t)3 * E,
                                           agg, (long long)E * Ho, Ho);
    packA(agg, agg_pk, Ho);
    gemm(agg_pk, W2P[3], sc, (float*)d_out, Ho, 64);  // N = 64, K = 64
  }
}
